// AMN_66554813218912
// MI455X (gfx1250) — matrix operand order repaired and checked by reading, not compiled, not run on hardware
//
#include <hip/hip_runtime.h>
#include <hip/hip_bf16.h>
#include <math.h>
#include <stdint.h>

// ---------------- problem dims ----------------
#define T_STEPS 100
#define TP      112          // T padded to 7 x 16 WMMA M-tiles
#define TT_TILES 7
#define U_UNITS 16
#define N_NEUR  1024
#define H_DIM   128

// ---------------- LIF constants ----------------
#define DT_C          1e-3f
#define TAU_MEM_INV_C (1.0f/20.0f)
#define TAU_SYN_INV_C (1.0f/10.0f)
#define V_LEAK_C      (-70.0f)
#define V_TH_C        (-55.0f)
#define V_RESET_C     (-70.0f)
#define DIRECT_W_C    1.5f

typedef __attribute__((ext_vector_type(16))) _Float16 v16h;
typedef __attribute__((ext_vector_type(8)))  _Float16 v8h;
typedef __attribute__((ext_vector_type(8)))  float    v8f;

__device__ __forceinline__ void pack8(v16h& v, int base, float4 x, float4 y) {
  v[base+0]=(_Float16)x.x; v[base+1]=(_Float16)x.y;
  v[base+2]=(_Float16)x.z; v[base+3]=(_Float16)x.w;
  v[base+4]=(_Float16)y.x; v[base+5]=(_Float16)y.y;
  v[base+6]=(_Float16)y.z; v[base+7]=(_Float16)y.w;
}

// ============================================================================
// K1: inp_cur[u,t,n] = sum_k spikes[t,k] * w_in[u,n,k]   (WMMA f16, f32 accum)
// One wave owns one 16-column n-tile and ALL 7 t-tiles: the B (weight)
// fragment is loaded once per K-step and reused by 7 independent WMMAs, so
// w_in streams from HBM exactly once (64 MB). Spike A-rows are tiny (400 KB)
// and stay L0/L2 resident.
// ============================================================================
__global__ __launch_bounds__(128) void snn_k1_input_gemm(
    const float* __restrict__ spikes, const float* __restrict__ w_in,
    float* __restrict__ inp_cur)
{
  const int lane = threadIdx.x & 31;
  const int wave = threadIdx.x >> 5;
  const int u    = blockIdx.x >> 4;        // 16 blocks per unit
  const int ng   = blockIdx.x & 15;
  const int ntile = ng * 4 + wave;
  const int half = lane >> 4;
  const int l    = lane & 15;
  const int col  = ntile * 16 + l;         // neuron index (B column)

  const float* brow = w_in + ((size_t)u * N_NEUR + col) * N_NEUR;

  v8f c[TT_TILES];
  #pragma unroll
  for (int tt = 0; tt < TT_TILES; ++tt) c[tt] = (v8f){};

  for (int k0 = 0; k0 < N_NEUR; k0 += 32) {
    // B fragment: elem e -> K = k0 + half*16 + e (16 contiguous floats/lane)
    v16h bm;
    const float4* q = (const float4*)(brow + k0 + half * 16);
    pack8(bm, 0, q[0], q[1]);
    pack8(bm, 8, q[2], q[3]);
    #pragma unroll
    for (int tt = 0; tt < TT_TILES; ++tt) {
      const int row = tt * 16 + l;         // time index (A row)
      v16h a;
      if (row < T_STEPS) {
        const float* arow = spikes + (size_t)row * N_NEUR;
        const float4* p0 = (const float4*)(arow + k0 + half * 16);
        const float4* p1 = (const float4*)(arow + k0 + half * 16 + 8);
        pack8(a, 0, p0[0], p0[1]);
        pack8(a, 8, p1[0], p1[1]);
      } else {
        #pragma unroll
        for (int e = 0; e < 16; ++e) a[e] = (_Float16)0.f;
      }
      c[tt] = __builtin_amdgcn_wmma_f32_16x16x32_f16(
          false, a, false, bm, (short)0, c[tt], false, false);
    }
  }
  // C layout: VGPR r holds M = r + half*8, N = l
  #pragma unroll
  for (int tt = 0; tt < TT_TILES; ++tt) {
    #pragma unroll
    for (int r = 0; r < 8; ++r) {
      const int tm = tt * 16 + r + half * 8;
      if (tm < T_STEPS)
        inp_cur[((size_t)u * T_STEPS + tm) * N_NEUR + col] = c[tt][r];
    }
  }
}

// ============================================================================
// K2: sequential LIF recurrence. One block (1024 threads) per unit; spike
// vector exchanged through LDS each step; w_rec[u] streams from L2 each step.
// Also writes z (f16) for the routing GEMM, zero-fills the T-pad rows, and
// accumulates per-neuron mean activity.
// ============================================================================
__global__ __launch_bounds__(1024) void snn_k2_lif(
    const float* __restrict__ w_rec, const float* __restrict__ inp_cur,
    _Float16* __restrict__ z_bf, float* __restrict__ activity)
{
  __shared__ float z_sh[N_NEUR];
  const int u = blockIdx.x;
  const int n = threadIdx.x;

  for (int t = T_STEPS; t < TP; ++t)                 // zero WMMA M-pad rows
    z_bf[((size_t)u * TP + t) * N_NEUR + n] = (_Float16)0.f;

  const float* wr = w_rec + ((size_t)u * N_NEUR + n) * N_NEUR;
  __builtin_prefetch(wr, 0, 1);                      // global_prefetch_b8

  float v = V_LEAK_C, cur = 0.f, z = 0.f, act = 0.f;
  for (int t = 0; t < T_STEPS; ++t) {
    z_sh[n] = z;                                     // previous-step spikes
    __syncthreads();
    const float v_dec = v + DT_C * TAU_MEM_INV_C * ((V_LEAK_C - v) + cur);
    const float i_dec = cur * (1.f - DT_C * TAU_SYN_INV_C);
    const float z_new = (v_dec - V_TH_C) > 0.f ? 1.f : 0.f;
    v = (1.f - z_new) * v_dec + z_new * V_RESET_C;

    float acc = 0.f;
    const float4* wr4 = (const float4*)wr;
    const float4* zs4 = (const float4*)z_sh;
    #pragma unroll 4
    for (int m = 0; m < N_NEUR / 4; ++m) {
      const float4 w4 = wr4[m]; const float4 s4 = zs4[m];
      acc = fmaf(w4.x, s4.x, acc); acc = fmaf(w4.y, s4.y, acc);
      acc = fmaf(w4.z, s4.z, acc); acc = fmaf(w4.w, s4.w, acc);
    }
    cur = i_dec + inp_cur[((size_t)u * T_STEPS + t) * N_NEUR + n] + acc;
    z_bf[((size_t)u * TP + t) * N_NEUR + n] = (_Float16)z_new;
    act += z_new;
    z = z_new;
    __syncthreads();                                 // protect z_sh WAR
  }
  activity[u * N_NEUR + n] = act * (1.f / T_STEPS);
}

// ============================================================================
// K3: coordinator MLP + Bernoulli routing matrix (tiny; one block).
// The uniform sample stands in for jax threefry(key=42): deterministic hash
// of the flat (i,j) index -> identical output on every call.
// ============================================================================
__global__ __launch_bounds__(256) void snn_k3_coord(
    const float* __restrict__ activity,
    const float* __restrict__ cw1, const float* __restrict__ cb1,
    const float* __restrict__ cw2, const float* __restrict__ cb2,
    float* __restrict__ conn)
{
  __shared__ float h_sh[H_DIM];
  const int tid = threadIdx.x;
  if (tid < H_DIM) {
    float acc = cb1[tid];
    const float* wrow = cw1 + (size_t)tid * (U_UNITS * N_NEUR);
    for (int k = 0; k < U_UNITS * N_NEUR; k += 4) {
      acc = fmaf(wrow[k+0], activity[k+0], acc);
      acc = fmaf(wrow[k+1], activity[k+1], acc);
      acc = fmaf(wrow[k+2], activity[k+2], acc);
      acc = fmaf(wrow[k+3], activity[k+3], acc);
    }
    h_sh[tid] = fmaxf(acc, 0.f);
  }
  __syncthreads();
  float logit = cb2[tid];
  const float* w2 = cw2 + tid * H_DIM;
  #pragma unroll 4
  for (int k = 0; k < H_DIM; ++k) logit = fmaf(w2[k], h_sh[k], logit);
  const float p = 1.f / (1.f + __expf(-logit));
  unsigned s = (unsigned)tid * 0x9E3779B9u ^ 0x2545F491u;  // threefry stand-in
  s ^= s >> 16; s *= 0x7FEB352Du;
  s ^= s >> 15; s *= 0x846CA68Bu;
  s ^= s >> 16;
  const float r = (float)(s >> 8) * (1.0f / 16777216.0f);
  conn[tid] = (r < p) ? 1.f : 0.f;
}

// ============================================================================
// K4: routed[j,t,n] = sum_i conn[i,j] * z[i,t,n]   (counts 0..16, exact in f16)
// ============================================================================
__global__ __launch_bounds__(256) void snn_k4_route(
    const _Float16* __restrict__ z_bf, const float* __restrict__ conn,
    _Float16* __restrict__ routed)
{
  const size_t idx = (size_t)blockIdx.x * 256 + threadIdx.x;   // j*(TP*N)+t*N+n
  const int j   = (int)(idx / ((size_t)TP * N_NEUR));
  const size_t rem = idx % ((size_t)TP * N_NEUR);
  float acc = 0.f;
  #pragma unroll
  for (int i = 0; i < U_UNITS; ++i)
    acc = fmaf(conn[i * U_UNITS + j],
               (float)z_bf[(size_t)i * TP * N_NEUR + rem], acc);
  routed[(size_t)j * TP * N_NEUR + rem] = (_Float16)acc;
}

// ============================================================================
// K5: out[t,m] = (1/U) * sum_j sum_n routed[j,t,n]*unit_w[j,n,m] + 1.5*spk[t,m]
// WMMA f16; unit loop accumulates into the C fragment (fused mean, no atomics).
// B columns are strided, so each wave stages a private 32x16 fp32 tile into
// LDS with GLOBAL_LOAD_ASYNC_TO_LDS_B128 (per-lane row copy; the shared
// INST_OFFSET advances global and LDS addresses together) and synchronizes
// with a single s_wait_asynccnt 0 — no block barriers needed since the tile
// is wave-private. fp32->f16 conversion happens on the fragment gather.
// ============================================================================
__global__ __launch_bounds__(128) void snn_k5_apply(
    const _Float16* __restrict__ routed, const float* __restrict__ unit_w,
    const float* __restrict__ spikes, float* __restrict__ out)
{
  __shared__ float btile[4][32 * 16];      // 8 KB, one 2 KB tile per wave
  const int lane = threadIdx.x & 31;
  const int wave = threadIdx.x >> 5;
  const int tt = blockIdx.x / 16;
  const int mg = blockIdx.x % 16;
  const int mtile = mg * 4 + wave;
  const int half = lane >> 4;
  const int l    = lane & 15;
  const int row  = tt * 16 + l;            // time index (pad rows read zeros)
  const int mcol = mtile * 16 + l;

  // per-lane LDS destination: low 32 bits of the generic address is the LDS
  // offset
  const unsigned lds_dst = (unsigned)(uintptr_t)&btile[wave][lane * 16];

  v8f c = {};
  for (int j = 0; j < U_UNITS; ++j) {
    const _Float16* ar = routed + ((size_t)j * TP + row) * N_NEUR;
    const float* wbase =
        unit_w + (size_t)j * N_NEUR * N_NEUR + (size_t)mtile * 16;
    for (int k0 = 0; k0 < N_NEUR; k0 += 32) {
      // lane kk async-copies unit_w[j, k0+kk, mtile*16 .. +16) -> LDS slot
      const float* gsrc = wbase + (size_t)(k0 + lane) * N_NEUR;
      asm volatile(
          "global_load_async_to_lds_b128 %0, %1, off\n\t"
          "global_load_async_to_lds_b128 %0, %1, off offset:16\n\t"
          "global_load_async_to_lds_b128 %0, %1, off offset:32\n\t"
          "global_load_async_to_lds_b128 %0, %1, off offset:48"
          :: "v"(lds_dst), "v"(gsrc) : "memory");
      asm volatile("s_wait_asynccnt 0" ::: "memory");

      v16h a, bm;
      const v8h a0 = *(const v8h*)(ar + k0 + half * 16);
      const v8h a1 = *(const v8h*)(ar + k0 + half * 16 + 8);
      #pragma unroll
      for (int e = 0; e < 8; ++e) { a[e] = a0[e]; a[8 + e] = a1[e]; }
      #pragma unroll
      for (int e = 0; e < 16; ++e)
        bm[e] = (_Float16)btile[wave][(half * 16 + e) * 16 + l];

      c = __builtin_amdgcn_wmma_f32_16x16x32_f16(
          false, a, false, bm, (short)0, c, false, false);
    }
  }
  #pragma unroll
  for (int r = 0; r < 8; ++r) {
    const int tm = tt * 16 + r + half * 8;
    if (tm < T_STEPS) {
      const size_t o = (size_t)tm * N_NEUR + mcol;
      out[o] = c[r] * (1.f / U_UNITS) + DIRECT_W_C * spikes[o];
    }
  }
}

// ============================================================================
extern "C" void kernel_launch(void* const* d_in, const int* in_sizes, int n_in,
                              void* d_out, int out_size, void* d_ws, size_t ws_size,
                              hipStream_t stream) {
  (void)in_sizes; (void)n_in; (void)out_size; (void)ws_size;
  const float* spikes = (const float*)d_in[0];   // [T,N]
  const float* w_in   = (const float*)d_in[1];   // [U,N,N]
  const float* w_rec  = (const float*)d_in[2];   // [U,N,N]
  const float* unit_w = (const float*)d_in[3];   // [U,N,N]
  const float* cw1    = (const float*)d_in[4];   // [H,U*N]
  const float* cb1    = (const float*)d_in[5];   // [H]
  const float* cw2    = (const float*)d_in[6];   // [U*U,H]
  const float* cb2    = (const float*)d_in[7];   // [U*U]
  float* out = (float*)d_out;                    // [T,N]

  // workspace layout (~14 MB, 16B-aligned offsets)
  char* ws = (char*)d_ws;
  float*    inp_cur  = (float*)(ws);                                   // 6,553,600 B
  _Float16* z_bf     = (_Float16*)(ws + 6553600);                      // 3,670,016 B
  _Float16* routed   = (_Float16*)(ws + 6553600 + 3670016);            // 3,670,016 B
  float*    activity = (float*)(ws + 6553600 + 2 * 3670016);           //    65,536 B
  float*    conn     = (float*)(ws + 6553600 + 2 * 3670016 + 65536);   //     1,024 B

  snn_k1_input_gemm<<<U_UNITS * 16, 128, 0, stream>>>(spikes, w_in, inp_cur);
  snn_k2_lif       <<<U_UNITS, 1024, 0, stream>>>(w_rec, inp_cur, z_bf, activity);
  snn_k3_coord     <<<1, 256, 0, stream>>>(activity, cw1, cb1, cw2, cb2, conn);
  snn_k4_route     <<<(U_UNITS * TP * N_NEUR) / 256, 256, 0, stream>>>(z_bf, conn, routed);
  snn_k5_apply     <<<TT_TILES * 16, 128, 0, stream>>>(routed, unit_w, spikes, out);
}
